// NeighborhoodAggr_52828097741150
// MI455X (gfx1250) — compile-verified, matrix operand order checked by reading, not run on hardware
//
#include <hip/hip_runtime.h>
#include <cstdint>

typedef __attribute__((ext_vector_type(16))) _Float16 v16h;
typedef __attribute__((ext_vector_type(8)))  float    v8f;
typedef int v4i __attribute__((vector_size(16)));   // matches builtin param type

#define DEG      32
#define IN_F     128
#define T_HALF   32       // T_F/2
#define T_F      64
#define OUT_F    128
#define CAT_F    256      // IN_F + HID
#define NSLOT    33       // 1 + DEG rows of tv/tp
#define M_PAD    48       // padded to 3 WMMA M-tiles
#define NB       16       // batch nodes per workgroup (proj GEMM M dim)
#define NTHREADS 256

#define AS_GLOBAL __attribute__((address_space(1)))
#define AS_LDS    __attribute__((address_space(3)))

// ---- CDNA5 async global->LDS copy (ASYNCcnt path), builtin or inline asm ----
__device__ __forceinline__ void async_copy_b128(const float* g, float* l) {
#if __has_builtin(__builtin_amdgcn_global_load_async_to_lds_b128)
    __builtin_amdgcn_global_load_async_to_lds_b128(
        (AS_GLOBAL v4i*)g, (AS_LDS v4i*)l, 0, 0);
#else
    unsigned lds_off = (unsigned)(unsigned long)(AS_LDS float*)l;
    asm volatile("global_load_async_to_lds_b128 %0, %1, off"
                 :: "v"(lds_off), "v"((unsigned long long)g) : "memory");
#endif
}

__device__ __forceinline__ void wait_async4() {
#if __has_builtin(__builtin_amdgcn_s_wait_asynccnt)
    __builtin_amdgcn_s_wait_asynccnt(4);
#else
    asm volatile("s_wait_asynccnt 0x4" ::: "memory");
#endif
}

__device__ __forceinline__ void wait_async0() {
#if __has_builtin(__builtin_amdgcn_s_wait_asynccnt)
    __builtin_amdgcn_s_wait_asynccnt(0);
#else
    asm volatile("s_wait_asynccnt 0x0" ::: "memory");
#endif
}

__launch_bounds__(NTHREADS, 1)
__global__ void tgat_aggr_kernel(const float* __restrict__ qmat,
                                 const float* __restrict__ kmat,
                                 const float* __restrict__ vmat,
                                 const int*   __restrict__ x0,
                                 const int*   __restrict__ neighbors,
                                 const float* __restrict__ times,
                                 const float* __restrict__ tq,
                                 const float* __restrict__ w_t2v,
                                 const float* __restrict__ b_t2v,
                                 const float* __restrict__ W_tp,
                                 const float* __restrict__ b_tp,
                                 const float* __restrict__ W_proj,
                                 const float* __restrict__ b_proj,
                                 float* __restrict__ out)
{
    __shared__ _Float16 Wh[IN_F][T_F];       // W_tp as f16 (B^T source), block-lifetime
    __shared__ _Float16 tvh[M_PAD][T_F];     // per-node A matrix (rows 33..47 zero)
    __shared__ float    tp[M_PAD][IN_F];     // per-node GEMM output + b_tp (rows >32 junk)
    __shared__ float    kn_s[DEG][IN_F];     // async-gathered kmat rows
    __shared__ float    vn_s[DEG][IN_F];     // async-gathered vmat rows
    __shared__ _Float16 cat_h[NB][CAT_F];    // [qn, h] per node, f16 (proj A matrix)
    __shared__ float    qn_s[IN_F];
    __shared__ float    btp_s[IN_F], bproj_s[IN_F];
    __shared__ float    w_s[T_HALF], bt_s[T_HALF];
    __shared__ int      nbr_all[NB][DEG];
    __shared__ int      x0_s[NB];
    __shared__ float    tt_all[NB][NSLOT];   // slot 0 = tq
    __shared__ float    maxv_all[NB];
    __shared__ float    logit_s[DEG], attn_s[DEG];
    __shared__ float    hpart[2][IN_F];

    const int tid  = threadIdx.x;
    const int lane = tid & 31;
    const int wv   = tid >> 5;               // wave id 0..7
    const int lmod = lane & 15;
    const int hsel = (lane < 16) ? 0 : 8;    // K-half select (ISA 16-bit A/B layout)
    const int b0   = blockIdx.x * NB;
    const float SQRT_HALF = 5.656854249492381f;   // sqrt(T_F/2)

    // ================= phase 0: block-lifetime staging =================
    if (tid < T_HALF) { w_s[tid] = w_t2v[tid]; bt_s[tid] = b_t2v[tid]; }
    if (tid < IN_F)   { btp_s[tid] = b_tp[tid]; bproj_s[tid] = b_proj[tid]; }
    if (tid < NB)       x0_s[tid] = x0[b0 + tid];
    for (int j = tid; j < IN_F * T_F; j += NTHREADS)
        ((_Float16*)Wh)[j] = (_Float16)W_tp[j];
    for (int j = tid; j < NB * DEG; j += NTHREADS)
        nbr_all[j >> 5][j & 31] = neighbors[(long)(b0 + (j >> 5)) * DEG + (j & 31)];
    for (int j = tid; j < NB * NSLOT; j += NTHREADS) {
        const int i = j / NSLOT, s = j % NSLOT;
        tt_all[i][s] = (s == 0) ? tq[b0 + i] : times[(long)(b0 + i) * DEG + s - 1];
    }
    __syncthreads();
    if (tid < NB) {
        float m = tt_all[tid][0];
        #pragma unroll
        for (int s = 1; s < NSLOT; ++s) m = fmaxf(m, tt_all[tid][s]);
        maxv_all[tid] = m;
    }
    __syncthreads();

    // ================= per-node loop =================
    for (int i = 0; i < NB; ++i) {
        // -- issue async gathers (overlap with Time2Vec + tp GEMM):
        //    4 wave-instructions for kmat rows, then 4 for vmat rows
        #pragma unroll
        for (int k = 0; k < 4; ++k) {
            const int c = tid + k * NTHREADS;        // 1024 b128 chunks = 16 KB
            const int d = c >> 5, sub = c & 31;
            async_copy_b128(kmat + (long)nbr_all[i][d] * IN_F + sub * 4,
                            &kn_s[d][sub * 4]);
        }
        #pragma unroll
        for (int k = 0; k < 4; ++k) {
            const int c = tid + k * NTHREADS;
            const int d = c >> 5, sub = c & 31;
            async_copy_b128(vmat + (long)nbr_all[i][d] * IN_F + sub * 4,
                            &vn_s[d][sub * 4]);
        }

        // -- Time2Vec: tvh[j] = [sin, cos](delta*w + b) * sqrt(32), f16, padded
        const float mval = maxv_all[i];
        for (int idx = tid; idx < M_PAD * T_HALF; idx += NTHREADS) {
            const int j = idx >> 5, t = idx & 31;
            float s = 0.f, c = 0.f;
            if (j < NSLOT) {
                const float enc = (mval - tt_all[i][j]) * w_s[t] + bt_s[t];
                s = __sinf(enc) * SQRT_HALF;
                c = __cosf(enc) * SQRT_HALF;
            }
            tvh[j][t]          = (_Float16)s;
            tvh[j][t + T_HALF] = (_Float16)c;
        }
        __syncthreads();

        // -- tp = tvh[48x64] @ Wh^T (+b_tp): 3x8 tiles, wave takes {wv, wv+8, wv+16}
        //    epilogue stores are unconditional (tp padded to 48 rows) to keep
        //    everything convergent -> compiler merges into 2addr DS stores.
        #pragma unroll
        for (int t = 0; t < 3; ++t) {
            const int tile = wv + t * 8;
            const int tm = tile >> 3, tn = tile & 7;
            v8f acc = {};
            #pragma unroll
            for (int kk = 0; kk < 2; ++kk) {
                union { v16h v; unsigned u[8]; } A, B;
                #pragma unroll
                for (int e = 0; e < 8; ++e) {
                    const int kb = ((e < 4) ? 0 : 16) + hsel + (e & 3) * 2 + kk * 32;
                    A.u[e] = *(const unsigned*)&tvh[tm * 16 + lmod][kb];
                    B.u[e] = *(const unsigned*)&Wh[tn * 16 + lmod][kb];
                }
                acc = __builtin_amdgcn_wmma_f32_16x16x32_f16(
                          false, A.v, false, B.v, (short)0, acc, false, false);
            }
            #pragma unroll
            for (int r = 0; r < 8; ++r) {
                const int row = tm * 16 + r + ((lane < 16) ? 0 : 8);
                const int n   = tn * 16 + lmod;
                tp[row][n] = acc[r] + btp_s[n];
            }
        }
        __syncthreads();

        // -- qn = qmat[x0] + tp[0]
        if (tid < IN_F) qn_s[tid] = qmat[(long)x0_s[i] * IN_F + tid] + tp[0][tid];
        wait_async4();          // kmat rows landed; vmat still in flight
        __syncthreads();

        // -- masked logits: wave wv handles neighbors 4*wv .. 4*wv+3
        #pragma unroll
        for (int dd = 0; dd < 4; ++dd) {
            const int d = wv * 4 + dd;
            float part = 0.f;
            #pragma unroll
            for (int ff = 0; ff < 4; ++ff) {
                const int f = lane + ff * 32;
                part += qn_s[f] * (kn_s[d][f] + tp[1 + d][f]);
            }
            #pragma unroll
            for (int off = 16; off > 0; off >>= 1)
                part += __shfl_xor(part, off, 32);
            if (lane == 0)
                logit_s[d] = (tt_all[i][1 + d] <= tt_all[i][0]) ? part : -1e9f;
        }
        __syncthreads();

        // -- softmax over 32 neighbors on wave 0 (lane == neighbor)
        if (wv == 0) {
            const float x = logit_s[lane];
            float m = x;
            #pragma unroll
            for (int off = 16; off > 0; off >>= 1)
                m = fmaxf(m, __shfl_xor(m, off, 32));
            const float e = __expf(x - m);
            float s = e;
            #pragma unroll
            for (int off = 16; off > 0; off >>= 1)
                s += __shfl_xor(s, off, 32);
            attn_s[lane] = e / s;
        }
        wait_async0();          // vmat rows landed
        __syncthreads();

        // -- h = sum_d attn[d] * (vn[d] + tp[1+d]); d-range split by half
        {
            const int f    = tid & (IN_F - 1);
            const int half = tid >> 7;
            float acc = 0.f;
            for (int d = half * 16; d < half * 16 + 16; ++d)
                acc += attn_s[d] * (vn_s[d][f] + tp[1 + d][f]);
            hpart[half][f] = acc;
        }
        __syncthreads();
        if (tid < IN_F) {
            cat_h[i][tid]        = (_Float16)qn_s[tid];
            cat_h[i][IN_F + tid] = (_Float16)(hpart[0][tid] + hpart[1][tid]);
        }
        __syncthreads();
    }

    // ================= proj GEMM: out[16x128] = relu(cat[16x256] @ W_proj^T + b) ====
    // M=16 (1 tile), N=128 (8 tiles, 1 per wave), K=256 (8 WMMA steps).
    // B fragments converted f32->f16 straight from global W_proj rows (K-contiguous).
    {
        const int tn = wv;
        const int o  = tn * 16 + lmod;
        v8f acc = {};
        #pragma unroll
        for (int kk = 0; kk < 8; ++kk) {
            union { v16h v; unsigned u[8]; } A, B;
            #pragma unroll
            for (int e = 0; e < 8; ++e) {
                const int kb = ((e < 4) ? 0 : 16) + hsel + (e & 3) * 2 + kk * 32;
                A.u[e] = *(const unsigned*)&cat_h[lmod][kb];
                const float2 wp = *(const float2*)&W_proj[(long)o * CAT_F + kb];
                union { _Float16 h[2]; unsigned u; } p;
                p.h[0] = (_Float16)wp.x;
                p.h[1] = (_Float16)wp.y;
                B.u[e] = p.u;
            }
            acc = __builtin_amdgcn_wmma_f32_16x16x32_f16(
                      false, A.v, false, B.v, (short)0, acc, false, false);
        }
        #pragma unroll
        for (int r = 0; r < 8; ++r) {
            const int node = r + ((lane < 16) ? 0 : 8);
            out[(long)(b0 + node) * OUT_F + o] = fmaxf(acc[r] + bproj_s[o], 0.f);
        }
    }
}

extern "C" void kernel_launch(void* const* d_in, const int* in_sizes, int n_in,
                              void* d_out, int out_size, void* d_ws, size_t ws_size,
                              hipStream_t stream) {
    (void)n_in; (void)out_size; (void)d_ws; (void)ws_size;
    const float* qmat      = (const float*)d_in[0];
    const float* kmat      = (const float*)d_in[1];
    const float* vmat      = (const float*)d_in[2];
    const int*   x0        = (const int*)  d_in[3];
    const int*   neighbors = (const int*)  d_in[4];
    const float* times     = (const float*)d_in[5];
    const float* tqv       = (const float*)d_in[6];
    const float* w_t2v     = (const float*)d_in[7];
    const float* b_t2v     = (const float*)d_in[8];
    const float* W_tp      = (const float*)d_in[9];
    const float* b_tp      = (const float*)d_in[10];
    const float* W_proj    = (const float*)d_in[11];
    const float* b_proj    = (const float*)d_in[12];
    float* out = (float*)d_out;

    const int batch = in_sizes[3];           // x0 has BATCH elements
    tgat_aggr_kernel<<<batch / NB, NTHREADS, 0, stream>>>(
        qmat, kmat, vmat, x0, neighbors, times, tqv,
        w_t2v, b_t2v, W_tp, b_tp, W_proj, b_proj, out);
}
